// MultiScaleMultiHeadAttention_28192165331544
// MI455X (gfx1250) — matrix operand order repaired and checked by reading, not compiled, not run on hardware
//
#include <hip/hip_runtime.h>
#include <hip/hip_bf16.h>

typedef __attribute__((ext_vector_type(16))) _Float16 v16h;
typedef __attribute__((ext_vector_type(8)))  _Float16 v8h;
typedef __attribute__((ext_vector_type(8)))  float    v8f;

// Build a 16-half WMMA fragment from two contiguous 16-byte chunks.
__device__ inline v16h make_frag(const _Float16* lo, const _Float16* hi) {
    v8h a = *(const v8h*)lo;
    v8h b = *(const v8h*)hi;
    v16h r;
#pragma unroll
    for (int i = 0; i < 8; ++i) { r[i] = a[i]; r[i + 8] = b[i]; }
    return r;
}

__device__ inline float rowmax16(float v) {
#pragma unroll
    for (int m = 1; m < 16; m <<= 1) v = fmaxf(v, __shfl_xor(v, m, 16));
    return v;
}
__device__ inline float rowsum16(float v) {
#pragma unroll
    for (int m = 1; m < 16; m <<= 1) v += __shfl_xor(v, m, 16);
    return v;
}

// ---------------------------------------------------------------------------
// Weight prep: fp32 -> f16, QKV fused into one 768x256 matrix + 768 bias vec.
// ---------------------------------------------------------------------------
__global__ void prep_weights_kernel(const float* __restrict__ Wq,
                                    const float* __restrict__ Wk,
                                    const float* __restrict__ Wv,
                                    const float* __restrict__ bq,
                                    const float* __restrict__ bk,
                                    const float* __restrict__ bv,
                                    const float* __restrict__ Wfc,
                                    _Float16* __restrict__ wqkv,
                                    _Float16* __restrict__ wfc,
                                    float* __restrict__ bqkv) {
    int idx = blockIdx.x * blockDim.x + threadIdx.x;
    if (idx < 768 * 256) {
        int m = idx >> 8;
        float v = (m < 256) ? Wq[idx] : (m < 512) ? Wk[idx - 256 * 256]
                                                  : Wv[idx - 512 * 256];
        wqkv[idx] = (_Float16)v;
    }
    if (idx < 256 * 256) wfc[idx] = (_Float16)Wfc[idx];
    if (idx < 768)
        bqkv[idx] = (idx < 256) ? bq[idx] : (idx < 512) ? bk[idx - 256] : bv[idx - 512];
}

// ---------------------------------------------------------------------------
// Build token-major f16 activations per scale.
// jax bilinear downsample by 2^s (antialias=False) == avg of a 2x2 tap at
// row/col offset (2^(s-1) - 1). Pad region [ntok, pad) zeroed.
// ---------------------------------------------------------------------------
__global__ void build_x_kernel(const float* __restrict__ x,
                               _Float16* __restrict__ xh,
                               int s, int ntok, int pad, int Ws) {
    int idx = blockIdx.x * blockDim.x + threadIdx.x;
    if (idx >= pad * 256) return;
    int n = idx >> 8, c = idx & 255;
    float v = 0.0f;
    if (n < ntok) {
        int ys = n / Ws, xs = n % Ws;
        const float* base = x + c * 2304;
        if (s == 0) {
            v = base[ys * 48 + xs];
        } else {
            int st = 1 << s, off = (1 << (s - 1)) - 1;
            int r0 = ys * st + off, c0 = xs * st + off;
            v = 0.25f * (base[r0 * 48 + c0] + base[r0 * 48 + c0 + 1] +
                         base[(r0 + 1) * 48 + c0] + base[(r0 + 1) * 48 + c0 + 1]);
        }
    }
    xh[(size_t)n * 256 + c] = (_Float16)v;
}

// ---------------------------------------------------------------------------
// Generic WMMA GEMM, 2x2 register-blocked: out[m,n] = sum_k A[m,k]*B[n,k]+bias.
// A: [M][256] f16 row-major (weights).  B: [Npad][256] f16 token-major.
// One wave computes a 32x32 macro-tile (4 wmma per k-step, 8 b128 loads) --
// K=256 fully unrolled (32 WMMAs per wave).
// mode 0: QKV epilogue -> Q[n][m] / K[n][m-256] token-major, V[(m-512)][n]
//         transposed (so the PV B-fragment is contiguous).
// mode 1: FC epilogue  -> f_out[m][n] fp32 channel-major.
// Requires M % 32 == 0 and Npad % 32 == 0 (true for all call sites).
// ---------------------------------------------------------------------------
__global__ __launch_bounds__(256) void gemm_wmma_kernel(
    const _Float16* __restrict__ A, const _Float16* __restrict__ B,
    const float* __restrict__ bias, int M, int Npad, int mode,
    _Float16* __restrict__ q_out, _Float16* __restrict__ k_out,
    _Float16* __restrict__ v_out, float* __restrict__ f_out) {
    int wave = threadIdx.x >> 5;
    int lane = threadIdx.x & 31;
    int hf = lane >> 4;
    int ll = lane & 15;
    int gm = M >> 5, gn = Npad >> 5;        // 32x32 macro-tile groups
    int id = blockIdx.x * 8 + wave;
    if (id >= gm * gn) return;
    int tm = (id % gm) * 2, tn = (id / gm) * 2;

    const _Float16* arow0 = A + (size_t)(tm * 16 + ll) * 256;
    const _Float16* arow1 = arow0 + (size_t)16 * 256;
    const _Float16* brow0 = B + (size_t)(tn * 16 + ll) * 256;
    const _Float16* brow1 = brow0 + (size_t)16 * 256;

    v8f acc00 = {}, acc01 = {}, acc10 = {}, acc11 = {};
#pragma unroll
    for (int k0 = 0; k0 < 256; k0 += 32) {
        v16h af0 = make_frag(arow0 + k0 + 16 * hf, arow0 + k0 + 16 * hf + 8);
        v16h af1 = make_frag(arow1 + k0 + 16 * hf, arow1 + k0 + 16 * hf + 8);
        v16h bf0 = make_frag(brow0 + k0 + 16 * hf, brow0 + k0 + 16 * hf + 8);
        v16h bf1 = make_frag(brow1 + k0 + 16 * hf, brow1 + k0 + 16 * hf + 8);
        acc00 = __builtin_amdgcn_wmma_f32_16x16x32_f16(false, af0, false, bf0,
                                                       (short)0, acc00, false, false);
        acc01 = __builtin_amdgcn_wmma_f32_16x16x32_f16(false, af0, false, bf1,
                                                       (short)0, acc01, false, false);
        acc10 = __builtin_amdgcn_wmma_f32_16x16x32_f16(false, af1, false, bf0,
                                                       (short)0, acc10, false, false);
        acc11 = __builtin_amdgcn_wmma_f32_16x16x32_f16(false, af1, false, bf1,
                                                       (short)0, acc11, false, false);
    }

    auto store_tile = [&](const v8f& acc, int tmi, int tnj) {
        int n = tnj * 16 + ll;
#pragma unroll
        for (int r = 0; r < 8; ++r) {
            int m = tmi * 16 + r + 8 * hf;
            float v = acc[r] + bias[m];
            if (mode == 0) {
                if (m < 256)      q_out[(size_t)n * 256 + m] = (_Float16)v;
                else if (m < 512) k_out[(size_t)n * 256 + (m - 256)] = (_Float16)v;
                else              v_out[(size_t)(m - 512) * Npad + n] = (_Float16)v;
            } else {
                f_out[(size_t)m * Npad + n] = v;
            }
        }
    };
    store_tile(acc00, tm, tn);
    store_tile(acc01, tm, tn + 1);
    store_tile(acc10, tm + 1, tn);
    store_tile(acc11, tm + 1, tn + 1);
}

// ---------------------------------------------------------------------------
// Flash attention, one wave per (head, 16-query tile).  d=32 == WMMA K.
// Scores/softmax online over 32-key chunks; P staged via per-wave LDS tile
// to become the next A-fragment (same-wave DS ordering, no barrier needed).
// Next chunk's K/V rows are prefetched (global_prefetch_b8) to overlap the
// L2->L0 fill with the current chunk's softmax VALU work.
// ---------------------------------------------------------------------------
__global__ __launch_bounds__(256) void flash_attn_kernel(
    const _Float16* __restrict__ Qh, const _Float16* __restrict__ Kh,
    const _Float16* __restrict__ Vt, _Float16* __restrict__ Oh,
    int ntok, int pad) {
    __shared__ __align__(16) _Float16 pbuf[8][16][32];
    int wave = threadIdx.x >> 5;
    int lane = threadIdx.x & 31;
    int hf = lane >> 4;
    int ll = lane & 15;
    int qtiles = (ntok + 15) >> 4;
    int qt = blockIdx.x * 8 + wave;
    bool valid = qt < qtiles;
    if (!valid) qt = qtiles - 1;   // keep all waves convergent
    int h = blockIdx.y;
    int q0 = qt * 16;

    const _Float16* qrow = Qh + (size_t)(q0 + ll) * 256 + h * 32;
    v16h qfrag = make_frag(qrow + 16 * hf, qrow + 16 * hf + 8);

    float mrow[8], lrow[8];
    v8f o0 = {}, o1 = {};
#pragma unroll
    for (int r = 0; r < 8; ++r) { mrow[r] = -1e30f; lrow[r] = 0.0f; }

    int nchunks = (ntok + 31) >> 5;
    for (int ch = 0; ch < nchunks; ++ch) {
        int c0 = ch * 32;
        // --- S = Q K^T for 32 keys (two 16-key subtiles) ---
        const _Float16* kr0 = Kh + (size_t)(c0 + ll) * 256 + h * 32 + 16 * hf;
        const _Float16* kr1 = Kh + (size_t)(c0 + 16 + ll) * 256 + h * 32 + 16 * hf;
        v16h kf0 = make_frag(kr0, kr0 + 8);
        v16h kf1 = make_frag(kr1, kr1 + 8);
        v8f z = {};
        v8f s0 = __builtin_amdgcn_wmma_f32_16x16x32_f16(false, qfrag, false, kf0,
                                                        (short)0, z, false, false);
        v8f s1 = __builtin_amdgcn_wmma_f32_16x16x32_f16(false, qfrag, false, kf1,
                                                        (short)0, z, false, false);
        // Prefetch next chunk's K and V rows while softmax runs.
        if (ch + 1 < nchunks) {
            int c1 = c0 + 32;
            __builtin_prefetch(Kh + (size_t)(c1 + ll) * 256 + h * 32, 0, 0);
            __builtin_prefetch(Kh + (size_t)(c1 + 16 + ll) * 256 + h * 32, 0, 0);
            __builtin_prefetch(Vt + (size_t)(h * 32 + ll) * pad + c1, 0, 0);
            __builtin_prefetch(Vt + (size_t)(h * 32 + 16 + ll) * pad + c1, 0, 0);
        }
        bool mask0 = (c0 + ll) >= ntok;
        bool mask1 = (c0 + 16 + ll) >= ntok;
        // --- online softmax over the 32-key chunk ---
#pragma unroll
        for (int r = 0; r < 8; ++r) {
            float a0 = mask0 ? -1e30f : s0[r];
            float a1 = mask1 ? -1e30f : s1[r];
            float mx = rowmax16(fmaxf(a0, a1));
            float mnew = fmaxf(mrow[r], mx);
            float alpha = __expf(mrow[r] - mnew);
            float p0 = __expf(a0 - mnew);
            float p1 = __expf(a1 - mnew);
            float rs = rowsum16(p0 + p1);
            lrow[r] = lrow[r] * alpha + rs;
            mrow[r] = mnew;
            o0[r] *= alpha;
            o1[r] *= alpha;
            pbuf[wave][r + 8 * hf][ll]      = (_Float16)p0;
            pbuf[wave][r + 8 * hf][16 + ll] = (_Float16)p1;
        }
        // --- O += P V ---
        const _Float16* prow = &pbuf[wave][ll][0];
        v16h pfrag = make_frag(prow + 16 * hf, prow + 16 * hf + 8);
        const _Float16* vr0 = Vt + (size_t)(h * 32 + ll) * pad + c0 + 16 * hf;
        const _Float16* vr1 = Vt + (size_t)(h * 32 + 16 + ll) * pad + c0 + 16 * hf;
        v16h vf0 = make_frag(vr0, vr0 + 8);
        v16h vf1 = make_frag(vr1, vr1 + 8);
        o0 = __builtin_amdgcn_wmma_f32_16x16x32_f16(false, pfrag, false, vf0,
                                                    (short)0, o0, false, false);
        o1 = __builtin_amdgcn_wmma_f32_16x16x32_f16(false, pfrag, false, vf1,
                                                    (short)0, o1, false, false);
    }
    if (valid) {
#pragma unroll
        for (int r = 0; r < 8; ++r) {
            int tok = q0 + r + 8 * hf;
            if (tok < ntok) {
                float inv = 1.0f / lrow[r];
                Oh[(size_t)tok * 256 + h * 32 + ll]      = (_Float16)(o0[r] * inv);
                Oh[(size_t)tok * 256 + h * 32 + 16 + ll] = (_Float16)(o1[r] * inv);
            }
        }
    }
}

// ---------------------------------------------------------------------------
// Bilinear upsample each scale's y (fp32 [C][pad], channel-major) to 48x48
// and sum into out48.
// ---------------------------------------------------------------------------
__global__ void upsample_accum_kernel(const float* __restrict__ y0,
                                      const float* __restrict__ y1,
                                      const float* __restrict__ y2,
                                      const float* __restrict__ y3,
                                      float* __restrict__ out48) {
    int idx = blockIdx.x * blockDim.x + threadIdx.x;
    if (idx >= 256 * 2304) return;
    int c = idx / 2304, p = idx % 2304;
    int j = p / 48, i = p % 48;
    float acc = y0[(size_t)c * 2304 + p];
    const float* ys[3] = {y1, y2, y3};
    const int Wss[3]  = {24, 12, 6};
    const int pads[3] = {576, 160, 64};
#pragma unroll
    for (int s = 0; s < 3; ++s) {
        int Ws = Wss[s];
        float sc = (float)Ws / 48.0f;
        float u = (i + 0.5f) * sc - 0.5f;
        float w = (j + 0.5f) * sc - 0.5f;
        int i0 = (int)floorf(u), j0 = (int)floorf(w);
        float fx = u - (float)i0, fy = w - (float)j0;
        int i0c = max(0, min(Ws - 1, i0)), i1c = max(0, min(Ws - 1, i0 + 1));
        int j0c = max(0, min(Ws - 1, j0)), j1c = max(0, min(Ws - 1, j0 + 1));
        const float* yc = ys[s] + (size_t)c * pads[s];
        float v00 = yc[j0c * Ws + i0c], v01 = yc[j0c * Ws + i1c];
        float v10 = yc[j1c * Ws + i0c], v11 = yc[j1c * Ws + i1c];
        acc += (1.0f - fy) * ((1.0f - fx) * v00 + fx * v01) +
               fy * ((1.0f - fx) * v10 + fx * v11);
    }
    out48[idx] = acc;
}

// ---------------------------------------------------------------------------
// SE gating + residual + global LayerNorm pipeline.
// ---------------------------------------------------------------------------
__global__ void se_avg_kernel(const float* __restrict__ out48, float* __restrict__ avg) {
    __shared__ float sh[256];
    int c = blockIdx.x;
    float s = 0.0f;
    for (int p = threadIdx.x; p < 2304; p += 256) s += out48[(size_t)c * 2304 + p];
    sh[threadIdx.x] = s;
    __syncthreads();
    for (int st = 128; st > 0; st >>= 1) {
        if (threadIdx.x < st) sh[threadIdx.x] += sh[threadIdx.x + st];
        __syncthreads();
    }
    if (threadIdx.x == 0) avg[c] = sh[0] / 2304.0f;
}

__global__ void se_fc_kernel(const float* __restrict__ avg,
                             const float* __restrict__ w1, const float* __restrict__ b1,
                             const float* __restrict__ a1,
                             const float* __restrict__ w2, const float* __restrict__ b2,
                             float* __restrict__ cw) {
    __shared__ float sh_avg[256];
    __shared__ float sh_h[16];
    sh_avg[threadIdx.x] = avg[threadIdx.x];
    __syncthreads();
    if (threadIdx.x < 16) {
        float s = b1[threadIdx.x];
        for (int c = 0; c < 256; ++c) s += sh_avg[c] * w1[threadIdx.x * 256 + c];
        sh_h[threadIdx.x] = (s > 0.0f) ? s : a1[0] * s;
    }
    __syncthreads();
    float s = b2[threadIdx.x];
    for (int j = 0; j < 16; ++j) s += sh_h[j] * w2[threadIdx.x * 16 + j];
    cw[threadIdx.x] = 1.0f / (1.0f + __expf(-s));
}

__global__ void se_apply_kernel(const float* __restrict__ out48,
                                const float* __restrict__ cw,
                                const float* __restrict__ x,
                                float* __restrict__ out,
                                float* __restrict__ partials) {
    __shared__ float s1[256], s2[256];
    int idx = blockIdx.x * 256 + threadIdx.x;   // 2304 blocks == 589824 exactly
    int c = idx / 2304;
    float v = out48[idx] * cw[c] + x[idx];
    out[idx] = v;
    s1[threadIdx.x] = v;
    s2[threadIdx.x] = v * v;
    __syncthreads();
    for (int st = 128; st > 0; st >>= 1) {
        if (threadIdx.x < st) {
            s1[threadIdx.x] += s1[threadIdx.x + st];
            s2[threadIdx.x] += s2[threadIdx.x + st];
        }
        __syncthreads();
    }
    if (threadIdx.x == 0) {
        partials[blockIdx.x * 2]     = s1[0];
        partials[blockIdx.x * 2 + 1] = s2[0];
    }
}

__global__ void ln_stats_kernel(const float* __restrict__ partials, float* __restrict__ stats) {
    __shared__ float s1[256], s2[256];
    float a = 0.0f, b = 0.0f;
    for (int i = threadIdx.x; i < 2304; i += 256) {
        a += partials[i * 2];
        b += partials[i * 2 + 1];
    }
    s1[threadIdx.x] = a;
    s2[threadIdx.x] = b;
    __syncthreads();
    for (int st = 128; st > 0; st >>= 1) {
        if (threadIdx.x < st) {
            s1[threadIdx.x] += s1[threadIdx.x + st];
            s2[threadIdx.x] += s2[threadIdx.x + st];
        }
        __syncthreads();
    }
    if (threadIdx.x == 0) {
        const float N = 589824.0f;
        float mu = s1[0] / N;
        float var = s2[0] / N - mu * mu;
        stats[0] = mu;
        stats[1] = rsqrtf(var + 1e-5f);
    }
}

__global__ void ln_apply_kernel(float* __restrict__ out,
                                const float* __restrict__ stats,
                                const float* __restrict__ a2) {
    int idx = blockIdx.x * blockDim.x + threadIdx.x;
    if (idx >= 589824) return;
    float v = (out[idx] - stats[0]) * stats[1];
    out[idx] = (v > 0.0f) ? v : a2[0] * v;
}

// ---------------------------------------------------------------------------
extern "C" void kernel_launch(void* const* d_in, const int* in_sizes, int n_in,
                              void* d_out, int out_size, void* d_ws, size_t ws_size,
                              hipStream_t stream) {
    (void)in_sizes; (void)n_in; (void)out_size; (void)ws_size;
    const float* x   = (const float*)d_in[0];
    const float* Wq  = (const float*)d_in[1];
    const float* bq  = (const float*)d_in[2];
    const float* Wk  = (const float*)d_in[3];
    const float* bk  = (const float*)d_in[4];
    const float* Wv  = (const float*)d_in[5];
    const float* bv  = (const float*)d_in[6];
    const float* Wfc = (const float*)d_in[7];
    const float* bfc = (const float*)d_in[8];
    const float* w1  = (const float*)d_in[9];
    const float* b1  = (const float*)d_in[10];
    const float* a1  = (const float*)d_in[11];
    const float* w2  = (const float*)d_in[12];
    const float* b2  = (const float*)d_in[13];
    const float* a2  = (const float*)d_in[14];

    char* ws = (char*)d_ws;
    size_t off = 0;
    auto alloc = [&](size_t bytes) -> void* {
        void* p = (void*)(ws + off);
        off += (bytes + 255) & ~(size_t)255;
        return p;
    };

    const int ntok[4] = {2304, 576, 144, 36};
    const int padt[4] = {2304, 576, 160, 64};   // multiples of 32
    const int Wsz[4]  = {48, 24, 12, 6};

    _Float16* wqkv = (_Float16*)alloc(768 * 256 * 2);
    _Float16* wfc  = (_Float16*)alloc(256 * 256 * 2);
    float*    bqkv = (float*)alloc(768 * 4);

    _Float16 *xh[4], *qh[4], *kh[4], *vt[4], *oh[4];
    float* ybuf[4];
    for (int s = 0; s < 4; ++s) {
        size_t tok = (size_t)padt[s];
        xh[s]   = (_Float16*)alloc(tok * 256 * 2);
        qh[s]   = (_Float16*)alloc(tok * 256 * 2);
        kh[s]   = (_Float16*)alloc(tok * 256 * 2);
        vt[s]   = (_Float16*)alloc(256 * tok * 2);
        oh[s]   = (_Float16*)alloc(tok * 256 * 2);
        ybuf[s] = (float*)alloc(256 * tok * 4);
    }
    float* out48    = (float*)alloc((size_t)2304 * 256 * 4);
    float* avg      = (float*)alloc(256 * 4);
    float* cw       = (float*)alloc(256 * 4);
    float* partials = (float*)alloc(2304 * 2 * 4);
    float* stats    = (float*)alloc(2 * 4);

    // 1) Weight prep
    prep_weights_kernel<<<(768 * 256 + 255) / 256, 256, 0, stream>>>(
        Wq, Wk, Wv, bq, bk, bv, Wfc, wqkv, wfc, bqkv);

    // 2) Per-scale activations
    for (int s = 0; s < 4; ++s)
        build_x_kernel<<<(padt[s] * 256 + 255) / 256, 256, 0, stream>>>(
            x, xh[s], s, ntok[s], padt[s], Wsz[s]);

    // 3) Per-scale: fused QKV GEMM -> flash attention -> output FC GEMM
    for (int s = 0; s < 4; ++s) {
        int groups_qkv = (768 >> 5) * (padt[s] >> 5);    // 32x32 macro-tiles
        gemm_wmma_kernel<<<(groups_qkv + 7) / 8, 256, 0, stream>>>(
            wqkv, xh[s], bqkv, 768, padt[s], 0, qh[s], kh[s], vt[s], nullptr);

        int qtiles = (ntok[s] + 15) >> 4;
        dim3 grid((qtiles + 7) / 8, 8);
        flash_attn_kernel<<<grid, 256, 0, stream>>>(
            qh[s], kh[s], vt[s], oh[s], ntok[s], padt[s]);

        int groups_fc = (256 >> 5) * (padt[s] >> 5);
        gemm_wmma_kernel<<<(groups_fc + 7) / 8, 256, 0, stream>>>(
            wfc, oh[s], bfc, 256, padt[s], 1, nullptr, nullptr, nullptr, ybuf[s]);
    }

    // 4) Upsample + accumulate all scales
    upsample_accum_kernel<<<(256 * 2304 + 255) / 256, 256, 0, stream>>>(
        ybuf[0], ybuf[1], ybuf[2], ybuf[3], out48);

    // 5) SE gating + residual + LayerNorm + PReLU
    se_avg_kernel<<<256, 256, 0, stream>>>(out48, avg);
    se_fc_kernel<<<1, 256, 0, stream>>>(avg, w1, b1, a1, w2, b2, cw);
    se_apply_kernel<<<2304, 256, 0, stream>>>(out48, cw, x, (float*)d_out, partials);
    ln_stats_kernel<<<1, 256, 0, stream>>>(partials, stats);
    ln_apply_kernel<<<(589824 + 255) / 256, 256, 0, stream>>>((float*)d_out, stats, a2);
}
